// Decoder_27736898798262
// MI455X (gfx1250) — matrix operand order repaired and checked by reading, not compiled, not run on hardware
//
#include <hip/hip_runtime.h>
#include <hip/hip_bf16.h>
#include <math.h>

// ---------------------------------------------------------------------------
// LSTM LM decoder on MI455X (gfx1250), bf16 WMMA everywhere, fused softmax-NLL.
//   T=64, B=32, E=H=1024, 4H=4096, V=32000, ROWS=T*B=2048
// ---------------------------------------------------------------------------

typedef __attribute__((ext_vector_type(16))) __bf16 v16bf;
typedef __attribute__((ext_vector_type(8)))  float  v8f;

#define T_STEPS    64
#define BATCH      32
#define EMB        1024
#define HID        1024
#define GATES      4096
#define VOCAB      32000
#define ROWS       2048
#define KDIM       1024
#define CHUNK_COLS 256
#define NCHUNK     125   // 125*256 = 32000
#define OP_ROWS    64    // rows per out_partial block (4 waves x 16)

union FragBF { v16bf v; unsigned int u[8]; };
union AccF   { v8f  v; float        f[8]; };

// ---- WMMA fragment loaders (layouts per CDNA5 ISA 7.12.2) -----------------
// A: 16x32 bf16, row-major source [*, lda]. lane<16: K in {0..7,16..23}; else {8..15,24..31}
__device__ __forceinline__ v16bf load_a_frag(const __bf16* A, int lda, int row0,
                                             int kbase, int lane) {
    FragBF fr;
    const int half = lane >> 4;
    const int row  = row0 + (lane & 15);
    const __bf16* p = A + (size_t)row * lda + kbase;
#pragma unroll
    for (int v = 0; v < 8; ++v) {
        const int k0 = ((v >> 2) << 4) + (half << 3) + ((v & 3) << 1);
        fr.u[v] = *(const unsigned int*)(p + k0);   // packed K pair
    }
    return fr.v;
}

// B: 32x16 bf16 from W^T stored [N][K] row-major. lane<16: K in {0..7,16..23}; else {8..15,24..31}
__device__ __forceinline__ v16bf load_b_frag_T(const __bf16* Wt, int ldk, int n0,
                                               int kbase, int lane) {
    FragBF fr;
    const int half = lane >> 4;
    const int col  = n0 + (lane & 15);
    const __bf16* p = Wt + (size_t)col * ldk + kbase;
#pragma unroll
    for (int j = 0; j < 8; ++j)
        fr.u[j] = *(const unsigned int*)(p + ((j >> 2) << 4) + (half << 3) + ((j & 3) << 1));
    return fr.v;
}

// ---- fp32 -> bf16 transposed weight conversion ----------------------------
__global__ void cvt_transpose_bf16(const float* __restrict__ W, __bf16* __restrict__ Wt,
                                   int K, int N) {
    long long idx = (long long)blockIdx.x * blockDim.x + threadIdx.x;
    if (idx >= (long long)K * N) return;
    int n = (int)(idx / K);
    int k = (int)(idx % K);
    Wt[idx] = (__bf16)W[(long long)k * N + n];
}

// ---- embedding gather -> bf16 ---------------------------------------------
__global__ void gather_embed(const int* __restrict__ tok, const float* __restrict__ emb,
                             __bf16* __restrict__ X) {
    int idx = blockIdx.x * blockDim.x + threadIdx.x;   // ROWS*EMB
    if (idx >= ROWS * EMB) return;
    int r = idx >> 10, e = idx & 1023;
    X[idx] = (__bf16)emb[(long long)tok[r] * EMB + e];
}

// ---- x_proj GEMM: X[2048,1024] x Wih^T -> xg[2048,4096] (+bias) -----------
__global__ void __launch_bounds__(32)
gemm_xproj(const __bf16* __restrict__ X, const __bf16* __restrict__ WihT,
           const float* __restrict__ bias, float* __restrict__ xg) {
    const int lane  = threadIdx.x;
    const int nbase = blockIdx.x * 64;
    const int mbase = blockIdx.y * 16;
    AccF acc[4];
#pragma unroll
    for (int i = 0; i < 4; ++i)
#pragma unroll
        for (int g = 0; g < 8; ++g) acc[i].f[g] = 0.0f;

    for (int kk = 0; kk < KDIM; kk += 32) {
        v16bf a = load_a_frag(X, EMB, mbase, kk, lane);
#pragma unroll
        for (int nt = 0; nt < 4; ++nt) {
            v16bf b = load_b_frag_T(WihT, KDIM, nbase + nt * 16, kk, lane);
            acc[nt].v = __builtin_amdgcn_wmma_f32_16x16x32_bf16(
                false, a, false, b, (short)0, acc[nt].v, false, false);
        }
    }
    const int half = lane >> 4, nlo = lane & 15;
#pragma unroll
    for (int nt = 0; nt < 4; ++nt) {
        const int col = nbase + nt * 16 + nlo;
        const float bv = bias[col];
#pragma unroll
        for (int g = 0; g < 8; ++g) {
            const int row = mbase + half * 8 + g;
            xg[(size_t)row * GATES + col] = acc[nt].f[g] + bv;
        }
    }
}

// ---- recurrent GEMM: gates = xg_t + h[32,1024] x Whh^T --------------------
__global__ void __launch_bounds__(32)
lstm_gemm(const __bf16* __restrict__ h, const __bf16* __restrict__ WhhT,
          const float* __restrict__ xg_t, float* __restrict__ gates) {
    const int lane  = threadIdx.x;
    const int nbase = blockIdx.x * 64;
    AccF acc[2][4];
#pragma unroll
    for (int m = 0; m < 2; ++m)
#pragma unroll
        for (int i = 0; i < 4; ++i)
#pragma unroll
            for (int g = 0; g < 8; ++g) acc[m][i].f[g] = 0.0f;

    for (int kk = 0; kk < KDIM; kk += 32) {
        v16bf a0 = load_a_frag(h, HID, 0,  kk, lane);
        v16bf a1 = load_a_frag(h, HID, 16, kk, lane);
#pragma unroll
        for (int nt = 0; nt < 4; ++nt) {
            v16bf b = load_b_frag_T(WhhT, KDIM, nbase + nt * 16, kk, lane);
            acc[0][nt].v = __builtin_amdgcn_wmma_f32_16x16x32_bf16(
                false, a0, false, b, (short)0, acc[0][nt].v, false, false);
            acc[1][nt].v = __builtin_amdgcn_wmma_f32_16x16x32_bf16(
                false, a1, false, b, (short)0, acc[1][nt].v, false, false);
        }
    }
    const int half = lane >> 4, nlo = lane & 15;
#pragma unroll
    for (int m = 0; m < 2; ++m)
#pragma unroll
        for (int nt = 0; nt < 4; ++nt) {
            const int col = nbase + nt * 16 + nlo;
#pragma unroll
            for (int g = 0; g < 8; ++g) {
                const int row = m * 16 + half * 8 + g;
                gates[(size_t)row * GATES + col] =
                    acc[m][nt].f[g] + xg_t[(size_t)row * GATES + col];
            }
        }
}

// ---- elementwise LSTM cell update -----------------------------------------
__global__ void lstm_cell(const float* __restrict__ gates, float* __restrict__ c,
                          __bf16* __restrict__ h, __bf16* __restrict__ hs_t) {
    int idx = blockIdx.x * blockDim.x + threadIdx.x;  // BATCH*HID = 32768
    if (idx >= BATCH * HID) return;
    int b = idx >> 10, j = idx & 1023;
    const float* gr = gates + (size_t)b * GATES;
    float gi = gr[j], gf = gr[HID + j], gg = gr[2 * HID + j], go = gr[3 * HID + j];
    float si = 1.0f / (1.0f + __expf(-gi));
    float sf = 1.0f / (1.0f + __expf(-gf));
    float so = 1.0f / (1.0f + __expf(-go));
    float cn = sf * c[idx] + si * tanhf(gg);
    float hv = so * tanhf(cn);
    c[idx] = cn;
    __bf16 hb = (__bf16)hv;
    h[idx] = hb;
    hs_t[idx] = hb;
}

// ---- output GEMM + fused online softmax partials --------------------------
// Block: 4 waves x 64 rows, sharing a 128KB LDS A-panel; each wave owns one
// 16-row M-tile and sweeps the 256-col chunk in groups of 4 N-tiles.
__global__ void __launch_bounds__(128)
out_partial(const __bf16* __restrict__ hs, const __bf16* __restrict__ WoutT,
            const float* __restrict__ b_out, const int* __restrict__ gt,
            float* __restrict__ pmax, float* __restrict__ psum,
            float* __restrict__ gtlogit) {
    extern __shared__ char smem[];
    __bf16* Alds = (__bf16*)smem;              // OP_ROWS * KDIM bf16 = 128KB
    const int tid  = threadIdx.x;
    const int lane = tid & 31;
    const int wave = tid >> 5;
    const int rowblk  = blockIdx.x * OP_ROWS;
    const int colbase = blockIdx.y * CHUNK_COLS;

    {   // contiguous 128KB copy: rows are row-major with ld == KDIM
        const uint4* src = (const uint4*)(hs + (size_t)rowblk * KDIM);
        uint4* dst = (uint4*)Alds;
        for (int i = tid; i < (OP_ROWS * KDIM) / 8; i += 128) dst[i] = src[i];
    }
    __syncthreads();

    const __bf16* Aw = Alds + wave * 16 * KDIM;
    const int rowbase = rowblk + wave * 16;
    const int half = lane >> 4, nlo = lane & 15;

    float rmax[8], rsum[8];
    int   gtcol[8];
#pragma unroll
    for (int g = 0; g < 8; ++g) {
        rmax[g] = -INFINITY;
        rsum[g] = 0.0f;
        gtcol[g] = gt[rowbase + half * 8 + g];
    }

    for (int ng = 0; ng < CHUNK_COLS / 64; ++ng) {   // 4 groups of 4 N-tiles
        const int n0g = colbase + ng * 64;
        if (ng < CHUNK_COLS / 64 - 1) {
            // prefetch next group's B rows into cache (global_prefetch_b8)
            const __bf16* nf = WoutT + (size_t)(n0g + 64 + lane * 2) * KDIM;
            __builtin_prefetch(nf, 0, 1);
            __builtin_prefetch(nf + KDIM, 0, 1);
        }
        AccF acc[4];
#pragma unroll
        for (int i = 0; i < 4; ++i)
#pragma unroll
            for (int g = 0; g < 8; ++g) acc[i].f[g] = 0.0f;

        for (int kk = 0; kk < KDIM; kk += 32) {
            v16bf a = load_a_frag(Aw, KDIM, 0, kk, lane);
#pragma unroll
            for (int nt = 0; nt < 4; ++nt) {
                v16bf b = load_b_frag_T(WoutT, KDIM, n0g + nt * 16, kk, lane);
                acc[nt].v = __builtin_amdgcn_wmma_f32_16x16x32_bf16(
                    false, a, false, b, (short)0, acc[nt].v, false, false);
            }
        }
#pragma unroll
        for (int nt = 0; nt < 4; ++nt) {
            const int col = n0g + nt * 16 + nlo;
            const float bo = b_out[col];
#pragma unroll
            for (int g = 0; g < 8; ++g) {
                float val = acc[nt].f[g] + bo;
                if (col == gtcol[g]) gtlogit[rowbase + half * 8 + g] = val;
                // row-wise reductions across the 16 lanes of this half
                float tmax = val;
                for (int m = 1; m < 16; m <<= 1) tmax = fmaxf(tmax, __shfl_xor(tmax, m, 32));
                float mnew = fmaxf(rmax[g], tmax);
                float p = __expf(val - mnew);
                float tsum = p;
                for (int m = 1; m < 16; m <<= 1) tsum += __shfl_xor(tsum, m, 32);
                rsum[g] = rsum[g] * __expf(rmax[g] - mnew) + tsum;
                rmax[g] = mnew;
            }
        }
    }
    if (nlo == 0) {
#pragma unroll
        for (int g = 0; g < 8; ++g) {
            const int row = rowbase + half * 8 + g;
            pmax[(size_t)row * NCHUNK + blockIdx.y] = rmax[g];
            psum[(size_t)row * NCHUNK + blockIdx.y] = rsum[g];
        }
    }
}

// ---- final log-sum-exp combine + mean NLL ---------------------------------
__global__ void reduce_nll(const float* __restrict__ pmax, const float* __restrict__ psum,
                           const float* __restrict__ gtlogit, float* __restrict__ out) {
    __shared__ float red[256];
    const int tid = threadIdx.x;
    float local = 0.0f;
    for (int r = tid; r < ROWS; r += 256) {
        float M = -INFINITY;
        for (int c = 0; c < NCHUNK; ++c) M = fmaxf(M, pmax[(size_t)r * NCHUNK + c]);
        float S = 0.0f;
        for (int c = 0; c < NCHUNK; ++c)
            S += psum[(size_t)r * NCHUNK + c] * __expf(pmax[(size_t)r * NCHUNK + c] - M);
        local += (M + __logf(S)) - gtlogit[r];
    }
    red[tid] = local;
    __syncthreads();
    for (int s = 128; s > 0; s >>= 1) {
        if (tid < s) red[tid] += red[tid + s];
        __syncthreads();
    }
    if (tid == 0) out[0] = red[0] / (float)ROWS;
}

// ---------------------------------------------------------------------------
extern "C" void kernel_launch(void* const* d_in, const int* in_sizes, int n_in,
                              void* d_out, int out_size, void* d_ws, size_t ws_size,
                              hipStream_t stream) {
    (void)in_sizes; (void)n_in; (void)out_size; (void)ws_size;
    const int*   tok   = (const int*)d_in[0];
    const int*   gt    = (const int*)d_in[1];
    const float* emb   = (const float*)d_in[2];
    const float* W_ih  = (const float*)d_in[3];
    const float* W_hh  = (const float*)d_in[4];
    const float* bias  = (const float*)d_in[5];
    const float* W_out = (const float*)d_in[6];
    const float* b_out = (const float*)d_in[7];
    float* out = (float*)d_out;

    char* wsp = (char*)d_ws;
    auto alloc = [&](size_t bytes) -> char* {
        char* p = wsp;
        wsp += (bytes + 255) & ~(size_t)255;
        return p;
    };
    __bf16* WihT  = (__bf16*)alloc((size_t)GATES * KDIM * 2);   // [4096][1024]
    __bf16* WhhT  = (__bf16*)alloc((size_t)GATES * KDIM * 2);   // [4096][1024]
    __bf16* WoutT = (__bf16*)alloc((size_t)VOCAB * KDIM * 2);   // [32000][1024]
    __bf16* X     = (__bf16*)alloc((size_t)ROWS * EMB * 2);     // gathered emb
    float*  xg    = (float*) alloc((size_t)ROWS * GATES * 4);   // input proj
    __bf16* hs    = (__bf16*)alloc((size_t)ROWS * HID * 2);     // all hidden states
    __bf16* hbuf  = (__bf16*)alloc((size_t)BATCH * HID * 2);
    float*  cbuf  = (float*) alloc((size_t)BATCH * HID * 4);
    float*  gbuf  = (float*) alloc((size_t)BATCH * GATES * 4);
    float*  pmax  = (float*) alloc((size_t)ROWS * NCHUNK * 4);
    float*  psum  = (float*) alloc((size_t)ROWS * NCHUNK * 4);
    float*  gtl   = (float*) alloc((size_t)ROWS * 4);

    // weight conversions (transposed, bf16)
    {
        long long n1 = (long long)GATES * KDIM;
        cvt_transpose_bf16<<<(unsigned)((n1 + 255) / 256), 256, 0, stream>>>(W_ih, WihT, KDIM, GATES);
        cvt_transpose_bf16<<<(unsigned)((n1 + 255) / 256), 256, 0, stream>>>(W_hh, WhhT, KDIM, GATES);
        long long n2 = (long long)VOCAB * KDIM;
        cvt_transpose_bf16<<<(unsigned)((n2 + 255) / 256), 256, 0, stream>>>(W_out, WoutT, KDIM, VOCAB);
    }
    gather_embed<<<(ROWS * EMB + 255) / 256, 256, 0, stream>>>(tok, emb, X);

    gemm_xproj<<<dim3(GATES / 64, ROWS / 16), 32, 0, stream>>>(X, WihT, bias, xg);

    hipMemsetAsync(hbuf, 0, (size_t)BATCH * HID * 2, stream);
    hipMemsetAsync(cbuf, 0, (size_t)BATCH * HID * 4, stream);

    for (int t = 0; t < T_STEPS; ++t) {
        lstm_gemm<<<GATES / 64, 32, 0, stream>>>(hbuf, WhhT,
                                                 xg + (size_t)t * BATCH * GATES, gbuf);
        lstm_cell<<<(BATCH * HID + 255) / 256, 256, 0, stream>>>(
            gbuf, cbuf, hbuf, hs + (size_t)t * BATCH * HID);
    }

    out_partial<<<dim3(ROWS / OP_ROWS, NCHUNK), 128,
                  (size_t)OP_ROWS * KDIM * sizeof(__bf16), stream>>>(
        hs, WoutT, b_out, gt, pmax, psum, gtl);
    reduce_nll<<<1, 256, 0, stream>>>(pmax, psum, gtl, out);
}
